// DecoderHead_86766929313982
// MI455X (gfx1250) — matrix operand order repaired and checked by reading, not compiled, not run on hardware
//
#include <hip/hip_runtime.h>
#include <hip/hip_bf16.h>
#include <math.h>

// ---------------------------------------------------------------------------
// MAE decoder for MI455X (gfx1250): wave32 + v_wmma_f32_16x16x32_f16 GEMMs,
// flash-attention with WMMA (V pre-transposed for contiguous-K B operands),
// DPP-based softmax reductions (VALU, not DS), fused LN/GELU/residual.
// ---------------------------------------------------------------------------

typedef _Float16 half4  __attribute__((ext_vector_type(4)));
typedef _Float16 half8  __attribute__((ext_vector_type(8)));
typedef _Float16 half16 __attribute__((ext_vector_type(16)));
typedef float    floatx8 __attribute__((ext_vector_type(8)));

__device__ __forceinline__ floatx8 zero8() {
  floatx8 z = {0.f,0.f,0.f,0.f,0.f,0.f,0.f,0.f};
  return z;
}

__device__ __forceinline__ half16 combine16(half8 lo, half8 hi) {
  return __builtin_shufflevector(lo, hi, 0,1,2,3,4,5,6,7,8,9,10,11,12,13,14,15);
}

__device__ __forceinline__ floatx8 wmma_f16(half16 a, half16 b, floatx8 c) {
  // D = A(16x32 f16) * B(32x16 f16) + C(16x16 f32)
  return __builtin_amdgcn_wmma_f32_16x16x32_f16(false, a, false, b,
                                                (short)0, c, false, false);
}

// ---- DPP lane-xor permutes within a 16-lane row (VALU, no DS pipe) --------
__device__ __forceinline__ float dpp_xor1(float x) {
  return __int_as_float(
      __builtin_amdgcn_mov_dpp(__float_as_int(x), 0xB1, 0xF, 0xF, true));
}
__device__ __forceinline__ float dpp_xor2(float x) {
  return __int_as_float(
      __builtin_amdgcn_mov_dpp(__float_as_int(x), 0x4E, 0xF, 0xF, true));
}
__device__ __forceinline__ float dpp_xor4(float x) {
  // DPP8 permute [4,5,6,7,0,1,2,3] within each 8-lane group
  return __int_as_float(__builtin_amdgcn_mov_dpp8(
      __float_as_int(x),
      (4 | (5 << 3) | (6 << 6) | (7 << 9) | (0 << 12) | (1 << 15) |
       (2 << 18) | (3 << 21))));
}
__device__ __forceinline__ float dpp_xor8(float x) {
  // row_ror:8 within 16-lane row == swap 8-lane halves == lane ^ 8
  return __int_as_float(
      __builtin_amdgcn_mov_dpp(__float_as_int(x), 0x128, 0xF, 0xF, true));
}
__device__ __forceinline__ float row16_max(float x) {
  x = fmaxf(x, dpp_xor1(x));
  x = fmaxf(x, dpp_xor2(x));
  x = fmaxf(x, dpp_xor4(x));
  x = fmaxf(x, dpp_xor8(x));
  return x;
}
__device__ __forceinline__ float row16_sum(float x) {
  x += dpp_xor1(x);
  x += dpp_xor2(x);
  x += dpp_xor4(x);
  x += dpp_xor8(x);
  return x;
}

// A-operand (16x32, f16): lane l holds row (m0 + (l&15)); the
// 16 elements are K = {16g..16g+15} with g = l>>4 (matches B).
__device__ __forceinline__ half16 load_a_frag(const _Float16* __restrict__ A,
                                              int ldk, int row, int k, int lane) {
  const _Float16* p = A + (size_t)row * ldk + k + ((lane >> 4) << 4);
  half8 lo = *(const half8*)p;
  half8 hi = *(const half8*)(p + 8);
  return combine16(lo, hi);
}

// B-operand (32x16, f16): lane l holds B column n = (n0 + (l&15)) with 16
// contiguous K values starting at 16g. With C = A * W^T (W stored [N][K]),
// column n of B is row n of W, contiguous in K.
__device__ __forceinline__ half16 load_b_frag(const _Float16* __restrict__ W,
                                              int ldk, int ncol, int k, int lane) {
  const _Float16* p = W + (size_t)ncol * ldk + k + ((lane >> 4) << 4);
  half8 lo = *(const half8*)p;
  half8 hi = *(const half8*)(p + 8);
  return combine16(lo, hi);
}

// ---------------------------------------------------------------------------
// f32 -> f16 weight conversion
// ---------------------------------------------------------------------------
__global__ void cvt_f16_kernel(const float* __restrict__ in,
                               _Float16* __restrict__ out, int n) {
  int i = blockIdx.x * 256 + threadIdx.x;
  if (i < n) out[i] = (_Float16)in[i];
}

// ---------------------------------------------------------------------------
// Scatter: build x (B,2048,768) from visible tokens + mask_token.
// idx[n] = clip(cumsum(mask)-1, 0, 511); one block per batch row.
// ---------------------------------------------------------------------------
__global__ void __launch_bounds__(256) scatter_kernel(
    const float* __restrict__ vt, const int* __restrict__ mask,
    const float* __restrict__ mtok, float* __restrict__ x) {
  __shared__ int sidx[2048];
  int b = blockIdx.x, tid = threadIdx.x;
  for (int n = tid; n < 2048; n += 256)
    sidx[n] = (mask[b * 2048 + n] != 0) ? 1 : 0;
  __syncthreads();
  if (tid == 0) {
    int run = 0;
    for (int n = 0; n < 2048; n++) {
      run += sidx[n];
      int v = run - 1;
      if (v < 0) v = 0;
      if (v > 511) v = 511;
      sidx[n] = v;
    }
  }
  __syncthreads();
  for (int i = tid; i < 2048 * 768; i += 256) {
    int n = i / 768, d = i - n * 768;
    bool vis = mask[b * 2048 + n] != 0;
    x[(size_t)b * 2048 * 768 + i] =
        vis ? vt[((size_t)b * 512 + sidx[n]) * 768 + d] : mtok[d];
  }
}

// ---------------------------------------------------------------------------
// LayerNorm over D=768, emits f16 (feeds WMMA GEMMs). One block per row.
// ---------------------------------------------------------------------------
__global__ void __launch_bounds__(256) ln_f16_kernel(
    const float* __restrict__ x, const float* __restrict__ g,
    const float* __restrict__ be, _Float16* __restrict__ y) {
  __shared__ float red[256];
  int row = blockIdx.x, tid = threadIdx.x;
  const float* xr = x + (size_t)row * 768;
  float s = 0.f;
  for (int i = tid; i < 768; i += 256) s += xr[i];
  red[tid] = s; __syncthreads();
  for (int st = 128; st > 0; st >>= 1) {
    if (tid < st) red[tid] += red[tid + st];
    __syncthreads();
  }
  float mu = red[0] * (1.0f / 768.0f);
  __syncthreads();
  float s2 = 0.f;
  for (int i = tid; i < 768; i += 256) { float d = xr[i] - mu; s2 += d * d; }
  red[tid] = s2; __syncthreads();
  for (int st = 128; st > 0; st >>= 1) {
    if (tid < st) red[tid] += red[tid + st];
    __syncthreads();
  }
  float rstd = rsqrtf(red[0] * (1.0f / 768.0f) + 1e-5f);
  for (int i = tid; i < 768; i += 256)
    y[(size_t)row * 768 + i] = (_Float16)((xr[i] - mu) * rstd * g[i] + be[i]);
}

// ---------------------------------------------------------------------------
// V transpose: extract V (cols [1536,2304) of qkv, ld 2304) into
// Vt[b][h][96][2048] so the attention P*V B-operand is contiguous in K
// (tokens). One block per (128-token tile, head); LDS tile transpose.
// ---------------------------------------------------------------------------
#define VT_PAD 100  // 128 rows x 96 feats, padded to de-phase LDS banks
__global__ void __launch_bounds__(256) vtrans_kernel(
    const _Float16* __restrict__ qkv, _Float16* __restrict__ vt) {
  __shared__ _Float16 tile[128 * VT_PAD];
  int tid = threadIdx.x;
  int tok0 = blockIdx.x << 7;     // global token base (0..8191), 128 per tile
  int h = blockIdx.y;
  int b = tok0 >> 11;
  int tloc0 = tok0 & 2047;
  // Phase 1: coalesced half4 reads of 128x96 V slice
  for (int i = tid; i < 128 * 24; i += 256) {
    int r = i / 24, c4 = i % 24;
    half4 v = *(const half4*)(qkv + (size_t)(tok0 + r) * 2304 + 1536 + h * 96 + c4 * 4);
    *(half4*)(tile + r * VT_PAD + c4 * 4) = v;
  }
  __syncthreads();
  // Phase 2: write transposed, coalesced half8 along tokens
  for (int i = tid; i < 96 * 16; i += 256) {
    int f = i / 16, c8 = i % 16;
    half8 v;
    for (int e = 0; e < 8; e++) v[e] = tile[(c8 * 8 + e) * VT_PAD + f];
    *(half8*)(vt + ((size_t)(b * 8 + h) * 96 + f) * 2048 + tloc0 + c8 * 8) = v;
  }
}

// ---------------------------------------------------------------------------
// WMMA GEMM: C = epi(A(f16, MxK) * W(f16, NxK)^T + bias).
// Block = 256 threads (8 waves), block tile 128x128, wave tile 32x64
// (2x4 WMMA accumulators -> 8 wmma : 12 b128 loads per K-step).
// EPI: 0 = f32 out; 1 = f16 out; 2 = f16 out + exact GELU; 3 = f32 out + resid.
// ---------------------------------------------------------------------------
template <int EPI>
__global__ void __launch_bounds__(256) gemm_f16_kernel(
    const _Float16* __restrict__ A, const _Float16* __restrict__ W,
    const float* __restrict__ bias, float* __restrict__ outf,
    _Float16* __restrict__ outh, const float* __restrict__ resid,
    int M, int N, int K) {
  int tid = threadIdx.x;
  int lane = tid & 31, w = tid >> 5;
  int mwave = (blockIdx.x << 7) + ((w & 3) << 5);   // 128 rows/block, 32/wave
  int nwave = (blockIdx.y << 7) + ((w >> 2) << 6);  // 128 cols/block, 64/wave

  floatx8 c[2][4];
  for (int mi = 0; mi < 2; mi++)
    for (int nj = 0; nj < 4; nj++) c[mi][nj] = zero8();

  int rA0 = mwave + (lane & 15), rA1 = rA0 + 16;
  int nB = nwave + (lane & 15);

  for (int k = 0; k < K; k += 32) {
    half16 a0 = load_a_frag(A, K, rA0, k, lane);
    half16 a1 = load_a_frag(A, K, rA1, k, lane);
    half16 b0 = load_b_frag(W, K, nB,      k, lane);
    half16 b1 = load_b_frag(W, K, nB + 16, k, lane);
    half16 b2 = load_b_frag(W, K, nB + 32, k, lane);
    half16 b3 = load_b_frag(W, K, nB + 48, k, lane);
    c[0][0] = wmma_f16(a0, b0, c[0][0]);
    c[0][1] = wmma_f16(a0, b1, c[0][1]);
    c[0][2] = wmma_f16(a0, b2, c[0][2]);
    c[0][3] = wmma_f16(a0, b3, c[0][3]);
    c[1][0] = wmma_f16(a1, b0, c[1][0]);
    c[1][1] = wmma_f16(a1, b1, c[1][1]);
    c[1][2] = wmma_f16(a1, b2, c[1][2]);
    c[1][3] = wmma_f16(a1, b3, c[1][3]);
  }

  for (int mi = 0; mi < 2; mi++) {
    for (int nj = 0; nj < 4; nj++) {
      int n = nwave + (nj << 4) + (lane & 15);
      float bv = bias[n];
      for (int r = 0; r < 8; r++) {
        // C layout: VGPR r, lane l -> M = r + 8*(l>>4), N = l&15
        int m = mwave + (mi << 4) + r + ((lane >> 4) << 3);
        float v = c[mi][nj][r] + bv;
        if (EPI == 2) v = 0.5f * v * (1.0f + erff(v * 0.70710678f));
        size_t oidx = (size_t)m * N + n;
        if (EPI == 3) v += resid[oidx];
        if (EPI == 0 || EPI == 3) outf[oidx] = v;
        else                      outh[oidx] = (_Float16)v;
      }
    }
  }
}

// ---------------------------------------------------------------------------
// Flash attention, one wave per (b, h, 16-row q-tile). qkv f16 (ld 2304,
// q at +0, k at +768, head h at +h*96, dh = 96 = 3*32); V read from the
// pre-transposed Vt[b][h][96][2048] so both WMMA B-operands use clean b128
// contiguous-K loads. Online softmax with DPP 16-lane reductions; P goes
// through a private per-wave LDS region to re-layout into the A-operand.
// ---------------------------------------------------------------------------
__global__ void __launch_bounds__(128) attn_kernel(
    const _Float16* __restrict__ qkv, const _Float16* __restrict__ vt,
    _Float16* __restrict__ outp) {
  __shared__ _Float16 lds_p[4][16 * 32];  // per-wave P tile (16x32 f16)
  int lane = threadIdx.x & 31;
  int w = threadIdx.x >> 5;
  int b = blockIdx.z, h = blockIdx.y;
  int q0 = (blockIdx.x * 4 + w) << 4;
  const int LD = 2304;
  const float scale = 0.1020620726f;  // 1/sqrt(96)

  // Q fragments (A-operand) for K-dim chunks c = 0..2
  half16 qf[3];
  {
    size_t qrow = (size_t)(b * 2048 + q0 + (lane & 15)) * LD + h * 96;
    for (int c = 0; c < 3; c++) {
      const _Float16* p = qkv + qrow + c * 32 + ((lane >> 4) << 4);
      qf[c] = combine16(*(const half8*)p, *(const half8*)(p + 8));
    }
  }
  const _Float16* vhead = vt + (size_t)(b * 8 + h) * 96 * 2048;

  float mrow[8], lrow[8];
  floatx8 O[6];
  for (int r = 0; r < 8; r++) { mrow[r] = -1e30f; lrow[r] = 0.f; }
  for (int j = 0; j < 6; j++) O[j] = zero8();

  for (int k0 = 0; k0 < 2048; k0 += 32) {
    // S chunk: 16 q-rows x 32 keys (two 16x16 WMMA tiles, 3 K-steps each)
    floatx8 s[2] = {zero8(), zero8()};
    for (int j = 0; j < 2; j++) {
      size_t krow = (size_t)(b * 2048 + k0 + j * 16 + (lane & 15)) * LD + 768 + h * 96;
      for (int c = 0; c < 3; c++) {
        const _Float16* p = qkv + krow + c * 32 + ((lane >> 4) << 4);
        half16 kb = combine16(*(const half8*)p, *(const half8*)(p + 8));
        s[j] = wmma_f16(qf[c], kb, s[j]);
      }
    }
    // Online softmax. Row m = r + 8*(lane>>4) lives on 16-lane halves;
    // DPP xor-butterfly (1,2,4,8) reduces within the half on the VALU.
    float p0[8], p1[8], sc[8];
    for (int r = 0; r < 8; r++) {
      float s0 = s[0][r] * scale, s1 = s[1][r] * scale;
      float mx = row16_max(fmaxf(s0, s1));
      float mn = fmaxf(mrow[r], mx);
      float scr = __expf(mrow[r] - mn);
      float e0 = __expf(s0 - mn), e1 = __expf(s1 - mn);
      float rs = row16_sum(e0 + e1);
      lrow[r] = lrow[r] * scr + rs;
      mrow[r] = mn;
      p0[r] = e0; p1[r] = e1; sc[r] = scr;
    }
    for (int j = 0; j < 6; j++)
      for (int r = 0; r < 8; r++) O[j][r] *= sc[r];

    // Bounce P through LDS: C-layout -> A-operand layout (same-wave LDS
    // ops are in-order; private per-wave region, no barrier needed).
    for (int r = 0; r < 8; r++) {
      int row = r + ((lane >> 4) << 3);
      int col = lane & 15;
      lds_p[w][row * 32 + col]      = (_Float16)p0[r];
      lds_p[w][row * 32 + col + 16] = (_Float16)p1[r];
    }
    const _Float16* pp = &lds_p[w][(lane & 15) * 32 + ((lane >> 4) << 4)];
    half16 pf = combine16(*(const half8*)pp, *(const half8*)(pp + 8));

    // P (16x32) * V (32x96): B-operand from Vt, contiguous tokens -> b128
    for (int j = 0; j < 6; j++) {
      const _Float16* vp =
          vhead + (size_t)(j * 16 + (lane & 15)) * 2048 + k0 + ((lane >> 4) << 4);
      half16 vb = combine16(*(const half8*)vp, *(const half8*)(vp + 8));
      O[j] = wmma_f16(pf, vb, O[j]);
    }
  }

  for (int r = 0; r < 8; r++) lrow[r] = 1.0f / lrow[r];
  for (int j = 0; j < 6; j++) {
    for (int r = 0; r < 8; r++) {
      int m = q0 + r + ((lane >> 4) << 3);
      int n = h * 96 + j * 16 + (lane & 15);
      outp[(size_t)(b * 2048 + m) * 768 + n] = (_Float16)(O[j][r] * lrow[r]);
    }
  }
}

// ---------------------------------------------------------------------------
// Host-side orchestration
// ---------------------------------------------------------------------------
extern "C" void kernel_launch(void* const* d_in, const int* in_sizes, int n_in,
                              void* d_out, int out_size, void* d_ws, size_t ws_size,
                              hipStream_t stream) {
  (void)in_sizes; (void)n_in; (void)out_size; (void)ws_size;
  const float* vtok   = (const float*)d_in[0];
  const int*   mask   = (const int*)d_in[1];
  const float* mtok   = (const float*)d_in[2];
  const float* gamma1 = (const float*)d_in[3];
  const float* beta1  = (const float*)d_in[4];
  const float* Wqkv   = (const float*)d_in[5];
  const float* bqkv   = (const float*)d_in[6];
  const float* Wo     = (const float*)d_in[7];
  const float* bo     = (const float*)d_in[8];
  const float* gamma2 = (const float*)d_in[9];
  const float* beta2  = (const float*)d_in[10];
  const float* W1     = (const float*)d_in[11];
  const float* b1     = (const float*)d_in[12];
  const float* W2     = (const float*)d_in[13];
  const float* b2     = (const float*)d_in[14];
  const float* gn     = (const float*)d_in[15];
  const float* gb     = (const float*)d_in[16];
  const float* Wdec   = (const float*)d_in[17];
  const float* bdec   = (const float*)d_in[18];

  const int M = 4 * 2048;  // 8192 tokens
  // Workspace layout
  char* p = (char*)d_ws;
  float*    xbuf  = (float*)p;            p += (size_t)M * 768 * 4;   // residual stream
  _Float16* ybuf  = (_Float16*)p;         p += (size_t)M * 768 * 2;   // LN output
  _Float16* qkvb  = (_Float16*)p;         p += (size_t)M * 2304 * 2;  // QKV
  _Float16* attnb = (_Float16*)p;         p += (size_t)M * 768 * 2;   // attn out
  _Float16* hbuf  = (_Float16*)p;         p += (size_t)M * 3072 * 2;  // MLP hidden
  _Float16* vtbuf = (_Float16*)p;         p += (size_t)4 * 8 * 96 * 2048 * 2; // V^T
  _Float16* wq16  = (_Float16*)p;         p += (size_t)2 * 2304 * 768 * 2;
  _Float16* wo16  = (_Float16*)p;         p += (size_t)2 * 768 * 768 * 2;
  _Float16* w116  = (_Float16*)p;         p += (size_t)2 * 3072 * 768 * 2;
  _Float16* w216  = (_Float16*)p;         p += (size_t)2 * 768 * 3072 * 2;
  _Float16* wd16  = (_Float16*)p;         p += (size_t)768 * 768 * 2;

  auto cvt = [&](const float* src, _Float16* dst, int n) {
    cvt_f16_kernel<<<(n + 255) / 256, 256, 0, stream>>>(src, dst, n);
  };
  cvt(Wqkv, wq16, 2 * 2304 * 768);
  cvt(Wo,   wo16, 2 * 768 * 768);
  cvt(W1,   w116, 2 * 3072 * 768);
  cvt(W2,   w216, 2 * 768 * 3072);
  cvt(Wdec, wd16, 768 * 768);

  scatter_kernel<<<4, 256, 0, stream>>>(vtok, mask, mtok, xbuf);

  for (int i = 0; i < 2; i++) {
    // LN1 -> QKV (M=8192, K=768, N=2304), f16 out
    ln_f16_kernel<<<M, 256, 0, stream>>>(xbuf, gamma1 + i * 768, beta1 + i * 768, ybuf);
    gemm_f16_kernel<1><<<dim3(M / 128, 2304 / 128), 256, 0, stream>>>(
        ybuf, wq16 + (size_t)i * 2304 * 768, bqkv + i * 2304,
        nullptr, qkvb, nullptr, M, 2304, 768);
    // V transpose, then attention (4 waves/block, one wave per q-tile)
    vtrans_kernel<<<dim3(M / 128, 8), 256, 0, stream>>>(qkvb, vtbuf);
    attn_kernel<<<dim3(32, 8, 4), 128, 0, stream>>>(qkvb, vtbuf, attnb);
    // Out-proj + residual (f32 x in-place)
    gemm_f16_kernel<3><<<dim3(M / 128, 768 / 128), 256, 0, stream>>>(
        attnb, wo16 + (size_t)i * 768 * 768, bo + i * 768,
        xbuf, nullptr, xbuf, M, 768, 768);
    // LN2 -> MLP1 (GELU, f16 out) -> MLP2 (+residual)
    ln_f16_kernel<<<M, 256, 0, stream>>>(xbuf, gamma2 + i * 768, beta2 + i * 768, ybuf);
    gemm_f16_kernel<2><<<dim3(M / 128, 3072 / 128), 256, 0, stream>>>(
        ybuf, w116 + (size_t)i * 3072 * 768, b1 + i * 3072,
        nullptr, hbuf, nullptr, M, 3072, 768);
    gemm_f16_kernel<3><<<dim3(M / 128, 768 / 128), 256, 0, stream>>>(
        hbuf, w216 + (size_t)i * 768 * 3072, b2 + i * 768,
        xbuf, nullptr, xbuf, M, 768, 3072);
  }

  // Final LN + decoder head (f32 out -> d_out)
  ln_f16_kernel<<<M, 256, 0, stream>>>(xbuf, gn, gb, ybuf);
  gemm_f16_kernel<0><<<dim3(M / 128, 768 / 128), 256, 0, stream>>>(
      ybuf, wd16, bdec, (float*)d_out, nullptr, nullptr, M, 768, 768);
}
